// TemporalAttentionLayer_61641370632363
// MI455X (gfx1250) — matrix operand order repaired and checked by reading, not compiled, not run on hardware
//
#include <hip/hip_runtime.h>

#define B_  16
#define T_  1024
#define D_  512
#define H_  8
#define DH_ 64

typedef __bf16 bf16;
typedef bf16  v16bf __attribute__((ext_vector_type(16)));
typedef bf16  v8bf  __attribute__((ext_vector_type(8)));
typedef float v8f   __attribute__((ext_vector_type(8)));
typedef unsigned short u16;

__device__ __forceinline__ u16 f2bf(float f) {
    union { float f; unsigned u; } v; v.f = f;
    unsigned u = v.u;
    u += 0x7FFFu + ((u >> 16) & 1u);       // round-to-nearest-even
    return (u16)(u >> 16);
}

__device__ __forceinline__ v8f wmma_bf16(v16bf a, v16bf b, v8f c) {
    return __builtin_amdgcn_wmma_f32_16x16x32_bf16(false, a, false, b, (short)0, c,
                                                   false, false);
}

// A-matrix 16x32 bf16 tile per-lane load (contiguous k order used by this file, matched by B):
// lane: row = lane%16, kb = (lane/16)*16
// elements[0..7]  = A[row][kb .. kb+7]
// elements[8..15] = A[row][kb+8 .. kb+15]
// caller passes p = &A[row][kchunk + (lane>>4)*16]
__device__ __forceinline__ v16bf load_a(const u16* p) {
    v8bf lo = *(const v8bf*)(p);
    v8bf hi = *(const v8bf*)(p + 8);
    v16bf r;
#pragma unroll
    for (int i = 0; i < 8; ++i) { r[i] = lo[i]; r[i + 8] = hi[i]; }
    return r;
}

// B-matrix 32x16 bf16 tile: lane holds col n = lane%16, K = (lane/16)*16 + [0..15]
// B[k][n] == Wrow_n[k]  ->  one contiguous 32B load per lane.

// ---------------- elementwise prep ----------------
__global__ void __launch_bounds__(256)
addpos_kernel(const float* __restrict__ x, const float* __restrict__ pos,
              u16* __restrict__ h, int n) {
    int i = blockIdx.x * blockDim.x + threadIdx.x;
    if (i < n) {
        int td = i % (T_ * D_);
        h[i] = f2bf(x[i] + pos[td]);
    }
}

__global__ void __launch_bounds__(256)
cvt_kernel(const float* __restrict__ in, u16* __restrict__ out, int n) {
    int i = blockIdx.x * blockDim.x + threadIdx.x;
    if (i < n) out[i] = f2bf(in[i]);
}

// ---------------- fused QKV projection ----------------
// one wave = one (b, head, 16-row t-tile); A-tile reused by 12 WMMAs (q,k,v x 4 n-tiles).
// q,k row-major [B,H,T,DH]; v stored transposed [B,H,DH,T] for the later P*V B-matrix.
__global__ void __launch_bounds__(256)
qkv_kernel(const u16* __restrict__ h,
           const u16* __restrict__ Wq, const u16* __restrict__ Wk,
           const u16* __restrict__ Wv,
           const float* __restrict__ bq, const float* __restrict__ bk,
           const float* __restrict__ bv,
           u16* __restrict__ q, u16* __restrict__ k, u16* __restrict__ vt) {
    const int wave = (blockIdx.x * blockDim.x + threadIdx.x) >> 5;
    const int lane = threadIdx.x & 31;
    const int lrow = lane & 15, lhi = lane >> 4;
    const int tpb  = T_ / 16;
    const int b    = wave / (H_ * tpb);
    const int hd   = (wave / tpb) % H_;
    const int i0   = (wave % tpb) * 16;

    v8f accq[4], acck[4], accv[4];
#pragma unroll
    for (int n = 0; n < 4; ++n)
#pragma unroll
        for (int r = 0; r < 8; ++r) { accq[n][r] = 0.f; acck[n][r] = 0.f; accv[n][r] = 0.f; }

    const u16* hrow = h + ((size_t)b * T_ + (i0 + lrow)) * D_;
    const u16* wqh  = Wq + (size_t)hd * DH_ * D_;
    const u16* wkh  = Wk + (size_t)hd * DH_ * D_;
    const u16* wvh  = Wv + (size_t)hd * DH_ * D_;

    for (int kk = 0; kk < D_; kk += 32) {
        v16bf a = load_a(hrow + kk + lhi * 16);
#pragma unroll
        for (int n = 0; n < 4; ++n) {
            int e = n * 16 + lrow;
            size_t wo = (size_t)e * D_ + kk + lhi * 16;
            accq[n] = wmma_bf16(a, *(const v16bf*)(wqh + wo), accq[n]);
            acck[n] = wmma_bf16(a, *(const v16bf*)(wkh + wo), acck[n]);
            accv[n] = wmma_bf16(a, *(const v16bf*)(wvh + wo), accv[n]);
        }
    }

#pragma unroll
    for (int n = 0; n < 4; ++n) {
        int e = n * 16 + lrow;
        float biq = bq[hd * DH_ + e], bik = bk[hd * DH_ + e], biv = bv[hd * DH_ + e];
#pragma unroll
        for (int r = 0; r < 8; ++r) {
            int t = i0 + r + lhi * 8;
            size_t qi = (((size_t)b * H_ + hd) * T_ + t) * DH_ + e;
            q[qi] = f2bf(accq[n][r] + biq);
            k[qi] = f2bf(acck[n][r] + bik);
            size_t vi = (((size_t)b * H_ + hd) * DH_ + e) * T_ + t;
            vt[vi] = f2bf(accv[n][r] + biv);
        }
    }
}

// ---------------- causal flash attention ----------------
// one wave = one (b, head, 32-row q group = 2 tiles). Both q tiles share every
// K/V block load (halves L2 streaming traffic vs 1 tile/wave). Online softmax
// with row stats kept in the C-tile layout (VGPR r <-> row r+8*(lane/16)).
__global__ void __launch_bounds__(256)
flash_kernel(const u16* __restrict__ q, const u16* __restrict__ kmat,
             const u16* __restrict__ vt,
             float* __restrict__ outf, u16* __restrict__ outb) {
    __shared__ u16 plds[8][16 * 32];      // per-wave 16x32 bf16 P staging tile
    const int wib  = threadIdx.x >> 5;
    const int wave = (blockIdx.x * blockDim.x + threadIdx.x) >> 5;
    const int lane = threadIdx.x & 31;
    const int lrow = lane & 15, lhi = lane >> 4;
    const int gpb  = T_ / 32;             // 32-row groups per (b,h)
    const int b    = wave / (H_ * gpb);
    const int hd   = (wave / gpb) % H_;
    const int i0   = (wave % gpb) * 32;   // multiple of 32

    const u16* qbase = q    + ((size_t)b * H_ + hd) * T_ * DH_;
    const u16* kbase = kmat + ((size_t)b * H_ + hd) * T_ * DH_;
    const u16* vbase = vt   + ((size_t)b * H_ + hd) * DH_ * T_;

    // Q A-tiles: 2 q-tiles x 2 e-chunks, loaded once
    v16bf qa[2][2];
#pragma unroll
    for (int g = 0; g < 2; ++g) {
        qa[g][0] = load_a(qbase + (size_t)(i0 + g * 16 + lrow) * DH_ + 0  + lhi * 16);
        qa[g][1] = load_a(qbase + (size_t)(i0 + g * 16 + lrow) * DH_ + 32 + lhi * 16);
    }

    v8f o[2][4];
    float m[2][8], l[2][8];
#pragma unroll
    for (int g = 0; g < 2; ++g) {
#pragma unroll
        for (int n = 0; n < 4; ++n)
#pragma unroll
            for (int r = 0; r < 8; ++r) o[g][n][r] = 0.f;
#pragma unroll
        for (int r = 0; r < 8; ++r) { m[g][r] = -3.0e38f; l[g][r] = 0.f; }
    }

    const float LOG2E = 1.4426950408889634f;
    const float SCALE = 0.125f;                  // 1/sqrt(DH)
    const float NEGV  = -4294967295.0f;          // reference mask value

    for (int j0 = 0; j0 <= i0 + 31; j0 += 32) {
        // ---- K B-tiles (2 jsub x 2 e-chunks) and V B-tiles (4 n-tiles): one load,
        //      shared by both q tiles.
        const u16* kp0 = kbase + (size_t)(j0 + lrow) * DH_ + lhi * 16;
        const u16* kp1 = kbase + (size_t)(j0 + 16 + lrow) * DH_ + lhi * 16;
        v16bf kb00 = *(const v16bf*)(kp0);
        v16bf kb01 = *(const v16bf*)(kp0 + 32);
        v16bf kb10 = *(const v16bf*)(kp1);
        v16bf kb11 = *(const v16bf*)(kp1 + 32);
        v16bf vb[4];
#pragma unroll
        for (int n = 0; n < 4; ++n)
            vb[n] = *(const v16bf*)(vbase + (size_t)(n * 16 + lrow) * T_ + j0 + lhi * 16);

#pragma unroll
        for (int g = 0; g < 2; ++g) {
            // ---- S = Q*K^T : two 16x16 C tiles
            v8f s0, s1;
#pragma unroll
            for (int r = 0; r < 8; ++r) { s0[r] = 0.f; s1[r] = 0.f; }
            s0 = wmma_bf16(qa[g][0], kb00, s0);
            s0 = wmma_bf16(qa[g][1], kb01, s0);
            s1 = wmma_bf16(qa[g][0], kb10, s1);
            s1 = wmma_bf16(qa[g][1], kb11, s1);

            // ---- scale + causal mask + online softmax update
            float sc0[8], sc1[8], mnew[8];
#pragma unroll
            for (int r = 0; r < 8; ++r) {
                int row = i0 + g * 16 + r + lhi * 8;
                int c0 = j0 + lrow, c1 = j0 + 16 + lrow;
                float a0 = (c0 <= row) ? s0[r] * SCALE : NEGV;
                float a1 = (c1 <= row) ? s1[r] * SCALE : NEGV;
                sc0[r] = a0; sc1[r] = a1;
                float mx = fmaxf(a0, a1);
                mx = fmaxf(mx, __shfl_xor(mx, 1));
                mx = fmaxf(mx, __shfl_xor(mx, 2));
                mx = fmaxf(mx, __shfl_xor(mx, 4));
                mx = fmaxf(mx, __shfl_xor(mx, 8));   // row max in 16-lane half
                mnew[r] = fmaxf(m[g][r], mx);
            }
#pragma unroll
            for (int r = 0; r < 8; ++r) {
                float p0 = __builtin_exp2f((sc0[r] - mnew[r]) * LOG2E);
                float p1 = __builtin_exp2f((sc1[r] - mnew[r]) * LOG2E);
                float rs = p0 + p1;
                rs += __shfl_xor(rs, 1); rs += __shfl_xor(rs, 2);
                rs += __shfl_xor(rs, 4); rs += __shfl_xor(rs, 8);
                float fac = __builtin_exp2f((m[g][r] - mnew[r]) * LOG2E);
                l[g][r] = l[g][r] * fac + rs;
                m[g][r] = mnew[r];
#pragma unroll
                for (int n = 0; n < 4; ++n) o[g][n][r] *= fac;
                // stash P (C-layout -> row-major LDS tile)
                plds[wib][(r + lhi * 8) * 32 + lrow]      = f2bf(p0);
                plds[wib][(r + lhi * 8) * 32 + 16 + lrow] = f2bf(p1);
            }
            // LDS is in-order per wave; reload P in A-matrix layout
            v16bf pa = load_a(&plds[wib][lrow * 32 + lhi * 16]);

            // ---- O += P * V
#pragma unroll
            for (int n = 0; n < 4; ++n)
                o[g][n] = wmma_bf16(pa, vb[n], o[g][n]);
        }
    }

    // ---- normalize and write concat-head output (f32 + bf16 copies)
#pragma unroll
    for (int g = 0; g < 2; ++g)
#pragma unroll
        for (int n = 0; n < 4; ++n)
#pragma unroll
            for (int r = 0; r < 8; ++r) {
                int t = i0 + g * 16 + r + lhi * 8;
                float val = o[g][n][r] / l[g][r];
                size_t oi = ((size_t)b * T_ + t) * D_ + hd * DH_ + n * 16 + lrow;
                outf[oi] = val;
                outb[oi] = f2bf(val);
            }
}

// ---------------- output projection + bias + inner residual ----------------
// one wave = 32 rows x 64 cols; each B-tile load feeds 2 WMMAs (2 row tiles).
__global__ void __launch_bounds__(256)
out_proj_kernel(const u16* __restrict__ ob, const u16* __restrict__ Wcb,
                const float* __restrict__ bc, const float* __restrict__ of,
                float* __restrict__ y) {
    const int wave = (blockIdx.x * blockDim.x + threadIdx.x) >> 5;
    const int lane = threadIdx.x & 31;
    const int lrow = lane & 15, lhi = lane >> 4;
    const int cgs  = D_ / 64;                    // 8 column groups
    const int rt   = wave / cgs;
    const int cg   = wave % cgs;
    const int row0 = rt * 32;                    // row in flattened [B*T]
    const int n0   = cg * 64;

    v8f acc[2][4];
#pragma unroll
    for (int g = 0; g < 2; ++g)
#pragma unroll
        for (int n = 0; n < 4; ++n)
#pragma unroll
            for (int r = 0; r < 8; ++r) acc[g][n][r] = 0.f;

    const u16* arow0 = ob + (size_t)(row0 + lrow) * D_;
    const u16* arow1 = ob + (size_t)(row0 + 16 + lrow) * D_;
    for (int kk = 0; kk < D_; kk += 32) {
        v16bf a0 = load_a(arow0 + kk + lhi * 16);
        v16bf a1 = load_a(arow1 + kk + lhi * 16);
#pragma unroll
        for (int n = 0; n < 4; ++n) {
            int e = n0 + n * 16 + lrow;
            v16bf wb = *(const v16bf*)(Wcb + (size_t)e * D_ + kk + lhi * 16);
            acc[0][n] = wmma_bf16(a0, wb, acc[0][n]);
            acc[1][n] = wmma_bf16(a1, wb, acc[1][n]);
        }
    }
#pragma unroll
    for (int g = 0; g < 2; ++g)
#pragma unroll
        for (int n = 0; n < 4; ++n) {
            int e = n0 + n * 16 + lrow;
            float bias = bc[e];
#pragma unroll
            for (int r = 0; r < 8; ++r) {
                size_t idx = (size_t)(row0 + g * 16 + r + lhi * 8) * D_ + e;
                y[idx] = acc[g][n][r] + bias + of[idx];   // y + out residual
            }
        }
}

extern "C" void kernel_launch(void* const* d_in, const int* in_sizes, int n_in,
                              void* d_out, int out_size, void* d_ws, size_t ws_size,
                              hipStream_t stream) {
    const float* x   = (const float*)d_in[0];
    const float* Wq  = (const float*)d_in[1];
    const float* bq  = (const float*)d_in[2];
    const float* Wk  = (const float*)d_in[3];
    const float* bk  = (const float*)d_in[4];
    const float* Wv  = (const float*)d_in[5];
    const float* bv  = (const float*)d_in[6];
    const float* pos = (const float*)d_in[7];
    const float* Wc  = (const float*)d_in[8];
    const float* bc  = (const float*)d_in[9];
    float* y = (float*)d_out;

    char*  ws  = (char*)d_ws;
    size_t off = 0;
    auto alloc = [&](size_t bytes) -> void* {
        void* p = ws + off;
        off += (bytes + 255) & ~(size_t)255;
        return p;
    };
    u16*   h_bf  = (u16*)alloc((size_t)B_ * T_ * D_ * 2);
    u16*   q_bf  = (u16*)alloc((size_t)B_ * H_ * T_ * DH_ * 2);
    u16*   k_bf  = (u16*)alloc((size_t)B_ * H_ * T_ * DH_ * 2);
    u16*   vt_bf = (u16*)alloc((size_t)B_ * H_ * DH_ * T_ * 2);
    u16*   o_bf  = (u16*)alloc((size_t)B_ * T_ * D_ * 2);
    float* o_f   = (float*)alloc((size_t)B_ * T_ * D_ * 4);
    u16*   Wq_bf = (u16*)alloc((size_t)H_ * DH_ * D_ * 2);
    u16*   Wk_bf = (u16*)alloc((size_t)H_ * DH_ * D_ * 2);
    u16*   Wv_bf = (u16*)alloc((size_t)H_ * DH_ * D_ * 2);
    u16*   Wc_bf = (u16*)alloc((size_t)D_ * D_ * 2);

    const int n1 = B_ * T_ * D_;
    addpos_kernel<<<(n1 + 255) / 256, 256, 0, stream>>>(x, pos, h_bf, n1);
    const int nw = H_ * DH_ * D_;
    cvt_kernel<<<(nw + 255) / 256, 256, 0, stream>>>(Wq, Wq_bf, nw);
    cvt_kernel<<<(nw + 255) / 256, 256, 0, stream>>>(Wk, Wk_bf, nw);
    cvt_kernel<<<(nw + 255) / 256, 256, 0, stream>>>(Wv, Wv_bf, nw);
    const int nc = D_ * D_;
    cvt_kernel<<<(nc + 255) / 256, 256, 0, stream>>>(Wc, Wc_bf, nc);

    const int waves_qkv = B_ * H_ * (T_ / 16);          // 8192 waves
    qkv_kernel<<<waves_qkv * 32 / 256, 256, 0, stream>>>(h_bf, Wq_bf, Wk_bf, Wv_bf,
                                                         bq, bk, bv, q_bf, k_bf, vt_bf);
    const int waves_fl = B_ * H_ * (T_ / 32);           // 4096 waves (32 rows each)
    flash_kernel<<<waves_fl * 32 / 256, 256, 0, stream>>>(q_bf, k_bf, vt_bf, o_f, o_bf);
    const int waves_op = (B_ * T_ / 32) * (D_ / 64);    // 4096 waves
    out_proj_kernel<<<waves_op * 32 / 256, 256, 0, stream>>>(o_bf, Wc_bf, bc, o_f, y);
}
